// Encoder_74363063763512
// MI455X (gfx1250) — matrix operand order repaired and checked by reading, not compiled, not run on hardware
//
#include <hip/hip_runtime.h>
#include <hip/hip_bf16.h>
#include <stdint.h>

// ---------------------------------------------------------------------------
// Bidirectional ELU-RNN for MI455X (gfx1250, wave32, WMMA).
//
//   h_t = elu( [e_t | h_{t-1}] @ W^T + b ),  W: (1024, 1536)
//
// Design (see analysis): ~206 GFLOP total, HBM traffic trivial -> the problem
// is the 512-long serial GEMM chain per direction. We keep each workgroup's
// 64-column slice of W (bf16, 192 KB) resident in LDS for all 512 steps,
// accumulate in f32 via v_wmma_f32_16x16x32_bf16, and synchronize the 16
// workgroups of a direction each step with a device-scope arrive counter.
// ---------------------------------------------------------------------------

typedef __bf16 bf16;
typedef __attribute__((ext_vector_type(8)))  __bf16 v8bf;
typedef __attribute__((ext_vector_type(16))) __bf16 v16bf;
typedef __attribute__((ext_vector_type(8)))  float  v8f;

#define BATCH   64
#define SEQL    512
#define EMBD    512
#define HIDD    1024
#define KTOT    (EMBD + HIDD)      // 1536
#define VOCABN  32000
#define NWG_DIR 16                 // workgroups per direction
#define NCOL_WG 64                 // hidden columns owned per workgroup
#define LDS_BYTES (NCOL_WG * KTOT * 2)   // 196608 B of bf16 weights in LDS

// ---------------------------------------------------------------------------
// Workspace layout (bf16 counts):
//   Wbf : 2 * HIDD * KTOT              = 3,145,728   (6 MB)
//   X   : SEQL * BATCH * EMBD          = 16,777,216  (32 MB)   layout [t][b][k]
//   H   : 2 dirs * 2 parity * B * HIDD = 262,144     (0.5 MB)
//   bar : 2 * u32 arrive counters
// ---------------------------------------------------------------------------
#define WBF_ELEMS  (2 * HIDD * KTOT)
#define X_ELEMS    (SEQL * BATCH * EMBD)
#define H_ELEMS    (2 * 2 * BATCH * HIDD)

__global__ void init_kernel(bf16* __restrict__ H, unsigned* __restrict__ bar) {
    int i = blockIdx.x * blockDim.x + threadIdx.x;
    if (i < H_ELEMS) H[i] = (bf16)0.0f;
    if (i < 2) bar[i] = 0u;
}

__global__ void prep_weights_kernel(const float* __restrict__ W_r,
                                    const float* __restrict__ W_l,
                                    bf16* __restrict__ Wbf) {
    int i = blockIdx.x * blockDim.x + threadIdx.x;   // 0 .. WBF_ELEMS-1
    const int per = HIDD * KTOT;
    if (i < per)            Wbf[i] = (bf16)W_r[i];
    else if (i < 2 * per)   Wbf[i] = (bf16)W_l[i - per];
}

__global__ void gather_embed_kernel(const int* __restrict__ inp,
                                    const float* __restrict__ E,
                                    bf16* __restrict__ X) {
    long i = (long)blockIdx.x * blockDim.x + threadIdx.x;  // 2^24 threads
    int k = (int)(i & (EMBD - 1));
    int b = (int)((i >> 9) & (BATCH - 1));
    int t = (int)(i >> 15);
    unsigned tok = ((unsigned)inp[b * SEQL + t]) % (unsigned)VOCABN;
    X[i] = (bf16)E[(size_t)tok * EMBD + k];
}

__device__ __forceinline__ float elu1(float x) {
    return x > 0.0f ? x : (__expf(x) - 1.0f);
}

__device__ __forceinline__ v16bf cat16(v8bf lo, v8bf hi) {
    return __builtin_shufflevector(lo, hi, 0,1,2,3,4,5,6,7,8,9,10,11,12,13,14,15);
}

// Persistent scan kernel. Grid: 32 blocks of 256 threads (8 wave32).
// blockIdx.x / 16 = direction (0 = forward, 1 = backward),
// blockIdx.x % 16 = 64-column slice of the hidden dimension.
extern "C" __global__ void __launch_bounds__(256, 1)
rnn_scan_kernel(const bf16* __restrict__ Wbf,
                const bf16* __restrict__ X,
                bf16* __restrict__ Hbuf,
                unsigned* __restrict__ bar,
                const float* __restrict__ b_r,
                const float* __restrict__ b_l,
                float* __restrict__ out) {
    extern __shared__ bf16 Bs[];                   // [NCOL_WG][KTOT] bf16 = 192 KB

    const int tid    = threadIdx.x;
    const int dir    = blockIdx.x >> 4;
    const int wg     = blockIdx.x & 15;
    const int wgbase = wg * NCOL_WG;

    // ---- preload this WG's 64 rows of W (bf16, row-major [n][k]) into LDS ----
    {
        const uint4* src = (const uint4*)(Wbf + (size_t)dir * HIDD * KTOT
                                              + (size_t)wgbase * KTOT);
        uint4* dst = (uint4*)Bs;
        const int n16 = NCOL_WG * KTOT / 8;        // 12288 uint4's
        for (int i = tid; i < n16; i += 256) dst[i] = src[i];
    }
    __syncthreads();

    const int wave  = tid >> 5;
    const int lane  = tid & 31;
    const int laneN = lane & 15;
    const int hi    = lane >> 4;                   // which half of the wave
    const int hi8   = hi * 8;
    const int hi16  = hi * 16;
    const int mtile = wave & 3;                    // batch tile: rows [mtile*16, +16)
    const int ngrp  = wave >> 2;                   // column pair group (0..1)

    const int ncolL0 = ngrp * 32 + laneN;          // two 16-col tiles per wave
    const int ncolL1 = ncolL0 + 16;
    const int ncolG0 = wgbase + ncolL0;
    const int ncolG1 = wgbase + ncolL1;
    const int m      = mtile * 16 + laneN;         // A-matrix row (batch index)

    const float* bias = dir ? b_l : b_r;
    const float  bb0  = bias[ncolG0];
    const float  bb1  = bias[ncolG1];

    const bf16* BsRow0 = Bs + (size_t)ncolL0 * KTOT;
    const bf16* BsRow1 = Bs + (size_t)ncolL1 * KTOT;

    bf16* h0buf = Hbuf + (size_t)dir * 2 * BATCH * HIDD;
    unsigned* mybar = bar + dir;

    for (int s = 0; s < SEQL; ++s) {
        const int tphys = dir ? (SEQL - 1 - s) : s;
        const bf16* Xt    = X     + (size_t)tphys * BATCH * EMBD;
        const bf16* hprev = h0buf + (size_t)(s & 1) * BATCH * HIDD;
        bf16*       hnext = h0buf + (size_t)((s + 1) & 1) * BATCH * HIDD;

        // prefetch next step's embedding row into cache
        if (s + 1 < SEQL) {
            const int tnext = dir ? (SEQL - 2 - s) : (s + 1);
            __builtin_prefetch(X + (size_t)tnext * BATCH * EMBD
                                 + (size_t)m * EMBD + lane, 0, 1);
        }

        v8f acc0, acc1;
        #pragma unroll
        for (int i = 0; i < 8; ++i) { acc0[i] = bb0; acc1[i] = bb1; }

        // ---- K over embedding part: k in [0, 512), A rows from X[t] ----
        const bf16* ArowX = Xt + (size_t)m * EMBD;
        #pragma unroll 4
        for (int kt = 0; kt < EMBD / 32; ++kt) {
            const bf16* pa = ArowX + kt * 32 + hi16;
            v16bf a = cat16(*(const v8bf*)pa, *(const v8bf*)(pa + 8));
            const bf16* pb0 = BsRow0 + kt * 32 + hi16;
            const bf16* pb1 = BsRow1 + kt * 32 + hi16;
            v16bf b0 = cat16(*(const v8bf*)pb0, *(const v8bf*)(pb0 + 8));
            v16bf b1 = cat16(*(const v8bf*)pb1, *(const v8bf*)(pb1 + 8));
            acc0 = __builtin_amdgcn_wmma_f32_16x16x32_bf16(
                       false, a, false, b0, (short)0, acc0, false, false);
            acc1 = __builtin_amdgcn_wmma_f32_16x16x32_bf16(
                       false, a, false, b1, (short)0, acc1, false, false);
        }

        // ---- K over recurrent part: k in [512, 1536), A rows from h_{t-1} ----
        const bf16* ArowH = hprev + (size_t)m * HIDD;
        #pragma unroll 4
        for (int kt = 0; kt < HIDD / 32; ++kt) {
            const bf16* pa = ArowH + kt * 32 + hi16;
            v16bf a = cat16(*(const v8bf*)pa, *(const v8bf*)(pa + 8));
            const bf16* pb0 = BsRow0 + EMBD + kt * 32 + hi16;
            const bf16* pb1 = BsRow1 + EMBD + kt * 32 + hi16;
            v16bf b0 = cat16(*(const v8bf*)pb0, *(const v8bf*)(pb0 + 8));
            v16bf b1 = cat16(*(const v8bf*)pb1, *(const v8bf*)(pb1 + 8));
            acc0 = __builtin_amdgcn_wmma_f32_16x16x32_bf16(
                       false, a, false, b0, (short)0, acc0, false, false);
            acc1 = __builtin_amdgcn_wmma_f32_16x16x32_bf16(
                       false, a, false, b1, (short)0, acc1, false, false);
        }

        // ---- epilogue: elu, publish h_next (bf16) and output (f32) ----
        #pragma unroll
        for (int i = 0; i < 8; ++i) {
            const int brow = mtile * 16 + hi8 + i;        // C layout: M = i + 8*hi
            const float y0 = elu1(acc0[i]);
            const float y1 = elu1(acc1[i]);
            hnext[(size_t)brow * HIDD + ncolG0] = (bf16)y0;
            hnext[(size_t)brow * HIDD + ncolG1] = (bf16)y1;
            const size_t ob = ((size_t)brow * SEQL + tphys) * (2 * HIDD)
                            + (size_t)dir * HIDD;
            __builtin_nontemporal_store(y0, &out[ob + ncolG0]);
            __builtin_nontemporal_store(y1, &out[ob + ncolG1]);
        }

        // ---- device-scope step barrier across this direction's 16 WGs ----
        __syncthreads();
        if (tid == 0) {
            __threadfence();                               // release h_next
            atomicAdd(mybar, 1u);
            const unsigned target = (unsigned)(s + 1) * NWG_DIR;
            while (__hip_atomic_load(mybar, __ATOMIC_ACQUIRE,
                                     __HIP_MEMORY_SCOPE_AGENT) < target) {
                __builtin_amdgcn_s_sleep(2);
            }
        }
        __syncthreads();
        __threadfence();                                   // acquire others' h_next
    }
}

// ---------------------------------------------------------------------------
extern "C" void kernel_launch(void* const* d_in, const int* in_sizes, int n_in,
                              void* d_out, int out_size, void* d_ws, size_t ws_size,
                              hipStream_t stream) {
    const int*   inp = (const int*)  d_in[0];
    const float* E   = (const float*)d_in[1];
    const float* W_r = (const float*)d_in[2];
    const float* b_r = (const float*)d_in[3];
    const float* W_l = (const float*)d_in[4];
    const float* b_l = (const float*)d_in[5];
    float* out = (float*)d_out;

    bf16* Wbf = (bf16*)d_ws;
    bf16* X   = Wbf + WBF_ELEMS;
    bf16* H   = X + X_ELEMS;
    unsigned* bar = (unsigned*)(H + H_ELEMS);

    // Allow 192 KB dynamic LDS (WGP supports up to 320 KB/workgroup).
    (void)hipFuncSetAttribute((const void*)rnn_scan_kernel,
                              hipFuncAttributeMaxDynamicSharedMemorySize,
                              LDS_BYTES);

    init_kernel<<<(H_ELEMS + 255) / 256, 256, 0, stream>>>(H, bar);
    prep_weights_kernel<<<WBF_ELEMS / 256, 256, 0, stream>>>(W_r, W_l, Wbf);
    gather_embed_kernel<<<X_ELEMS / 256, 256, 0, stream>>>(inp, E, X);
    rnn_scan_kernel<<<2 * NWG_DIR, 256, LDS_BYTES, stream>>>(
        Wbf, X, H, bar, b_r, b_l, out);
}
